// SharedFusedMoE_36438502539323
// MI455X (gfx1250) — matrix operand order repaired and checked by reading, not compiled, not run on hardware
//
#include <hip/hip_runtime.h>
#include <cstdint>
#include <cstddef>

#define NUM_EXPERTS 16
#define TOPK 2

typedef __attribute__((ext_vector_type(16))) __bf16 v16bf;
typedef __attribute__((ext_vector_type(8)))  float  v8f;

union FragBF { v16bf v; uint4 q[2]; };

__device__ __forceinline__ unsigned short f2bf(float f) {
  unsigned u = __float_as_uint(f);
  u += 0x7FFFu + ((u >> 16) & 1u);          // round-to-nearest-even
  return (unsigned short)(u >> 16);
}

// ---------------------------------------------------------------------------
// WMMA fragment loaders from LDS tiles stored K-contiguous ([rows][32] bf16).
// A 16x32 bf16 layout: lane L (m=L%16, h=L/16): elems 0..7 -> k=h*8..h*8+7,
//                      elems 8..15 -> k=16+h*8..16+h*8+7.
// B 32x16 bf16 layout: lane L (n=L%16, h=L/16): elems 0..7 -> k=h*8..h*8+7,
//                      elems 8..15 -> k=16+h*8..16+h*8+7.
// ---------------------------------------------------------------------------
__device__ __forceinline__ v16bf frag_a(const unsigned short* As, int m) {
  int lane = (int)(threadIdx.x & 31);
  const unsigned short* p = As + (m + (lane & 15)) * 32 + ((lane >> 4) * 8);
  FragBF f;
  f.q[0] = *(const uint4*)(p);
  f.q[1] = *(const uint4*)(p + 16);
  return f.v;
}

__device__ __forceinline__ v16bf frag_b(const unsigned short* Bs, int n) {
  int lane = (int)(threadIdx.x & 31);
  const unsigned short* p = Bs + (n + (lane & 15)) * 32 + ((lane >> 4) * 8);
  FragBF f;
  f.q[0] = *(const uint4*)(p);
  f.q[1] = *(const uint4*)(p + 16);
  return f.v;
}

// ---------------------------------------------------------------------------
// fp32 -> bf16 conversion (grid-stride)
// ---------------------------------------------------------------------------
__global__ void k_convert(const float* __restrict__ src,
                          unsigned short* __restrict__ dst, long n) {
  long i = (long)blockIdx.x * blockDim.x + threadIdx.x;
  long stride = (long)gridDim.x * blockDim.x;
  for (; i < n; i += stride) dst[i] = f2bf(src[i]);
}

// ---------------------------------------------------------------------------
// Router: top-2 of softmax, renormalized == softmax over the two top logits.
// ---------------------------------------------------------------------------
__global__ void k_router(const float* __restrict__ logits,
                         int* __restrict__ ti, float* __restrict__ tw, int T) {
  int t = blockIdx.x * blockDim.x + threadIdx.x;
  if (t >= T) return;
  const float* l = logits + (size_t)t * NUM_EXPERTS;
  int i1 = 0; float m1 = l[0];
  for (int e = 1; e < NUM_EXPERTS; ++e) { float v = l[e]; if (v > m1) { m1 = v; i1 = e; } }
  int i2 = (i1 == 0) ? 1 : 0; float m2 = l[i2];
  for (int e = 0; e < NUM_EXPERTS; ++e) {
    if (e == i1) continue;
    float v = l[e]; if (v > m2) { m2 = v; i2 = e; }
  }
  float w1 = 1.0f / (1.0f + __expf(m2 - m1));
  ti[2 * t] = i1; ti[2 * t + 1] = i2;
  tw[2 * t] = w1; tw[2 * t + 1] = 1.0f - w1;
}

__global__ void k_zero_i32(int* p, int n) {
  int i = blockIdx.x * blockDim.x + threadIdx.x;
  if (i < n) p[i] = 0;
}

__global__ void k_count(const int* __restrict__ ti, int* __restrict__ counts, int T) {
  int t = blockIdx.x * blockDim.x + threadIdx.x;
  if (t >= T) return;
  atomicAdd(&counts[ti[2 * t]], 1);
  atomicAdd(&counts[ti[2 * t + 1]], 1);
}

__global__ void k_prefix(const int* __restrict__ counts, int* __restrict__ offs) {
  if (threadIdx.x == 0) {
    int s = 0;
    for (int e = 0; e < NUM_EXPERTS; ++e) { offs[e] = s; s += counts[e]; }
    offs[NUM_EXPERTS] = s;
  }
}

// Deterministic per-expert token compaction (block e scans tokens in order).
__global__ __launch_bounds__(256) void k_scatter(const int* __restrict__ ti,
                                                 const int* __restrict__ offs,
                                                 int* __restrict__ tlist,
                                                 int* __restrict__ tslot, int T) {
  int e = blockIdx.x;
  __shared__ int sb[256];
  int base = offs[e];
  int running = 0;
  int tid = (int)threadIdx.x;
  for (int c0 = 0; c0 < T; c0 += 256) {
    int t = c0 + tid;
    int k = -1;
    if (t < T) {
      if (ti[2 * t] == e) k = 0;
      else if (ti[2 * t + 1] == e) k = 1;
    }
    int flag = (k >= 0) ? 1 : 0;
    sb[tid] = flag;
    __syncthreads();
    for (int off = 1; off < 256; off <<= 1) {
      int v = sb[tid];
      if (tid >= off) v += sb[tid - off];
      __syncthreads();
      sb[tid] = v;
      __syncthreads();
    }
    int incl = sb[tid];
    int tot = sb[255];
    if (k >= 0) {
      int s = base + running + incl - 1;
      tlist[s] = t;
      tslot[2 * t + k] = s;
    }
    running += tot;
    __syncthreads();
  }
}

// ---------------------------------------------------------------------------
// SwiGLU GEMM: act = silu(X*Wg^T) * (X*Wu^T), bf16 in / bf16 out, f32 acc.
// Block tile 64x128(act), 8 waves (2x4), wave tile 32x32 for gate AND up.
// Double-buffered LDS ping-pong: stage K-slice i+1 while WMMAs eat slice i.
// counts==nullptr: shared-expert mode.  counts!=nullptr: routed (z=expert),
// A rows gathered through the per-expert token list.
// ---------------------------------------------------------------------------
__global__ __launch_bounds__(256) void k_swiglu(
    const unsigned short* __restrict__ X, int K,
    const unsigned short* __restrict__ W13, size_t wstride, int Nact,
    unsigned short* __restrict__ act,
    const int* __restrict__ counts, const int* __restrict__ offs,
    const int* __restrict__ tlist, int T) {
  int cnt = T;
  const int* list = nullptr;
  const unsigned short* W = W13;
  unsigned short* A_out = act;
  if (counts) {
    int e = blockIdx.z;
    cnt = counts[e];
    int seg = offs[e];
    W = W13 + (size_t)e * wstride;
    A_out = act + (size_t)seg * Nact;
    list = tlist + seg;
  }
  int m0 = blockIdx.y * 64;
  if (m0 >= cnt) return;
  int n0 = blockIdx.x * 128;

  __shared__ alignas(16) unsigned short As[2][64 * 32];
  __shared__ alignas(16) unsigned short Bg[2][128 * 32];
  __shared__ alignas(16) unsigned short Bu[2][128 * 32];

  int tid = (int)threadIdx.x, lane = tid & 31, wid = tid >> 5;
  int wm = (wid >> 2) * 32, wn = (wid & 3) * 32;

  v8f cg[2][2] = {};
  v8f cu[2][2] = {};

  int ar = tid >> 2;           // A tile row this thread stages (0..63)
  int ac = (tid & 3) * 8;      // k sub-offset (16B chunk)
  int br = tid >> 1;           // B tile row (0..127)
  int bc = (tid & 1) * 16;     // k sub-offset (32B chunk)

  const unsigned short* aPtr = nullptr;
  {
    int r = m0 + ar;
    if (r < cnt) {
      long arow = list ? (long)list[r] : (long)r;
      aPtr = X + arow * (size_t)K + ac;
    }
  }
  const unsigned short* gRow = W + (size_t)(n0 + br) * K + bc;
  const unsigned short* uRow = W + (size_t)(Nact + n0 + br) * K + bc;

  // prologue: stage K-slice 0 into buffer 0
  uint4 aR = make_uint4(0u, 0u, 0u, 0u);
  uint4 bg0, bg1, bu0, bu1;
  if (aPtr) aR = *(const uint4*)(aPtr);
  bg0 = *(const uint4*)(gRow);     bg1 = *(const uint4*)(gRow + 8);
  bu0 = *(const uint4*)(uRow);     bu1 = *(const uint4*)(uRow + 8);
  *(uint4*)(As[0] + ar * 32 + ac)     = aR;
  *(uint4*)(Bg[0] + br * 32 + bc)     = bg0;
  *(uint4*)(Bg[0] + br * 32 + bc + 8) = bg1;
  *(uint4*)(Bu[0] + br * 32 + bc)     = bu0;
  *(uint4*)(Bu[0] + br * 32 + bc + 8) = bu1;

  int niter = K / 32;
  for (int i = 0; i < niter; ++i) {
    __syncthreads();
    int cur = i & 1;
    bool more = (i + 1) < niter;
    if (more) {
      int k0 = (i + 1) * 32;
      aR = make_uint4(0u, 0u, 0u, 0u);
      if (aPtr) aR = *(const uint4*)(aPtr + k0);
      bg0 = *(const uint4*)(gRow + k0);     bg1 = *(const uint4*)(gRow + k0 + 8);
      bu0 = *(const uint4*)(uRow + k0);     bu1 = *(const uint4*)(uRow + k0 + 8);
    }
    for (int fm = 0; fm < 2; ++fm) {
      v16bf a = frag_a(As[cur], wm + fm * 16);
      for (int fn = 0; fn < 2; ++fn) {
        v16bf vbg = frag_b(Bg[cur], wn + fn * 16);
        v16bf vbu = frag_b(Bu[cur], wn + fn * 16);
        cg[fm][fn] = __builtin_amdgcn_wmma_f32_16x16x32_bf16(
            false, a, false, vbg, (short)0, cg[fm][fn], false, false);
        cu[fm][fn] = __builtin_amdgcn_wmma_f32_16x16x32_bf16(
            false, a, false, vbu, (short)0, cu[fm][fn], false, false);
      }
    }
    if (more) {
      int nb = cur ^ 1;
      *(uint4*)(As[nb] + ar * 32 + ac)     = aR;
      *(uint4*)(Bg[nb] + br * 32 + bc)     = bg0;
      *(uint4*)(Bg[nb] + br * 32 + bc + 8) = bg1;
      *(uint4*)(Bu[nb] + br * 32 + bc)     = bu0;
      *(uint4*)(Bu[nb] + br * 32 + bc + 8) = bu1;
    }
  }

  for (int fm = 0; fm < 2; ++fm)
    for (int fn = 0; fn < 2; ++fn)
      for (int v = 0; v < 8; ++v) {
        int r = m0 + wm + fm * 16 + ((lane >> 4) * 8) + v;
        if (r >= cnt) continue;
        int c = n0 + wn + fn * 16 + (lane & 15);
        float g = cg[fm][fn][v];
        float u = cu[fm][fn][v];
        float s = g / (1.0f + __expf(-g));
        A_out[(size_t)r * Nact + c] = f2bf(s * u);
      }
}

// ---------------------------------------------------------------------------
// Down GEMM: out = A * W2^T, bf16 in / f32 out.
// Block tile 64x256, 8 waves (2x4), wave tile 32x64 (2x4 frags): each A
// fragment feeds 4 WMMAs. Double-buffered LDS ping-pong as above.
// counts==nullptr: shared mode (out -> d_out). Routed: per-slot f32 buffer.
// ---------------------------------------------------------------------------
__global__ __launch_bounds__(256) void k_down(
    const unsigned short* __restrict__ Abase, int K,
    const unsigned short* __restrict__ W2, size_t wstride, int N,
    float* __restrict__ outb,
    const int* __restrict__ counts, const int* __restrict__ offs, int T) {
  int cnt = T;
  const unsigned short* A = Abase;
  const unsigned short* W = W2;
  float* out = outb;
  if (counts) {
    int e = blockIdx.z;
    cnt = counts[e];
    int seg = offs[e];
    A = Abase + (size_t)seg * K;
    W = W2 + (size_t)e * wstride;
    out = outb + (size_t)seg * N;
  }
  int m0 = blockIdx.y * 64;
  if (m0 >= cnt) return;
  int n0 = blockIdx.x * 256;

  __shared__ alignas(16) unsigned short As[2][64 * 32];
  __shared__ alignas(16) unsigned short Bs[2][256 * 32];

  int tid = (int)threadIdx.x, lane = tid & 31, wid = tid >> 5;
  int wm = (wid >> 2) * 32, wn = (wid & 3) * 64;

  v8f c[2][4] = {};

  int ar = tid >> 2;           // A tile row (0..63)
  int ac = (tid & 3) * 8;
  const unsigned short* aPtr = (m0 + ar < cnt)
      ? A + (size_t)(m0 + ar) * K + ac : nullptr;
  const unsigned short* bRow = W + (size_t)(n0 + tid) * K;  // one B row/thread

  // prologue: stage K-slice 0
  uint4 aR = make_uint4(0u, 0u, 0u, 0u);
  uint4 bR0, bR1, bR2, bR3;
  if (aPtr) aR = *(const uint4*)(aPtr);
  bR0 = *(const uint4*)(bRow);      bR1 = *(const uint4*)(bRow + 8);
  bR2 = *(const uint4*)(bRow + 16); bR3 = *(const uint4*)(bRow + 24);
  *(uint4*)(As[0] + ar * 32 + ac)  = aR;
  *(uint4*)(Bs[0] + tid * 32)      = bR0;
  *(uint4*)(Bs[0] + tid * 32 + 8)  = bR1;
  *(uint4*)(Bs[0] + tid * 32 + 16) = bR2;
  *(uint4*)(Bs[0] + tid * 32 + 24) = bR3;

  int niter = K / 32;
  for (int i = 0; i < niter; ++i) {
    __syncthreads();
    int cur = i & 1;
    bool more = (i + 1) < niter;
    if (more) {
      int k0 = (i + 1) * 32;
      aR = make_uint4(0u, 0u, 0u, 0u);
      if (aPtr) aR = *(const uint4*)(aPtr + k0);
      bR0 = *(const uint4*)(bRow + k0);      bR1 = *(const uint4*)(bRow + k0 + 8);
      bR2 = *(const uint4*)(bRow + k0 + 16); bR3 = *(const uint4*)(bRow + k0 + 24);
    }
    for (int fm = 0; fm < 2; ++fm) {
      v16bf a = frag_a(As[cur], wm + fm * 16);
      for (int fn = 0; fn < 4; ++fn) {
        v16bf b = frag_b(Bs[cur], wn + fn * 16);
        c[fm][fn] = __builtin_amdgcn_wmma_f32_16x16x32_bf16(
            false, a, false, b, (short)0, c[fm][fn], false, false);
      }
    }
    if (more) {
      int nb = cur ^ 1;
      *(uint4*)(As[nb] + ar * 32 + ac)  = aR;
      *(uint4*)(Bs[nb] + tid * 32)      = bR0;
      *(uint4*)(Bs[nb] + tid * 32 + 8)  = bR1;
      *(uint4*)(Bs[nb] + tid * 32 + 16) = bR2;
      *(uint4*)(Bs[nb] + tid * 32 + 24) = bR3;
    }
  }

  for (int fm = 0; fm < 2; ++fm)
    for (int fn = 0; fn < 4; ++fn)
      for (int v = 0; v < 8; ++v) {
        int r = m0 + wm + fm * 16 + ((lane >> 4) * 8) + v;
        if (r >= cnt) continue;
        int cc = n0 + wn + fn * 16 + (lane & 15);
        out[(size_t)r * N + cc] = c[fm][fn][v];
      }
}

// ---------------------------------------------------------------------------
// Combine: fused[t] = w0*y[slot(t,0)] + w1*y[slot(t,1)]  (no atomics)
// ---------------------------------------------------------------------------
__global__ void k_combine(const float* __restrict__ y, const int* __restrict__ tslot,
                          const float* __restrict__ tw, float* __restrict__ fused, int D) {
  int t = blockIdx.x;
  int s0 = tslot[2 * t], s1 = tslot[2 * t + 1];
  float w0 = tw[2 * t], w1 = tw[2 * t + 1];
  const float4* a = (const float4*)(y + (size_t)s0 * D);
  const float4* b = (const float4*)(y + (size_t)s1 * D);
  float4* o = (float4*)(fused + (size_t)t * D);
  for (int i = (int)threadIdx.x; i < D / 4; i += (int)blockDim.x) {
    float4 va = a[i], vb = b[i];
    float4 r;
    r.x = w0 * va.x + w1 * vb.x;
    r.y = w0 * va.y + w1 * vb.y;
    r.z = w0 * va.z + w1 * vb.z;
    r.w = w0 * va.w + w1 * vb.w;
    o[i] = r;
  }
}

// ---------------------------------------------------------------------------
extern "C" void kernel_launch(void* const* d_in, const int* in_sizes, int n_in,
                              void* d_out, int out_size, void* d_ws, size_t ws_size,
                              hipStream_t stream) {
  const float* hidden = (const float*)d_in[0];
  const float* logits = (const float*)d_in[1];
  const float* w13    = (const float*)d_in[2];
  const float* w2     = (const float*)d_in[3];
  const float* w13s   = (const float*)d_in[4];
  const float* w2s    = (const float*)d_in[5];
  float* out = (float*)d_out;

  const long E = NUM_EXPERTS;
  const long T  = (long)in_sizes[1] / E;
  const long D  = (long)in_sizes[0] / T;
  const long I  = (long)in_sizes[3] / (E * D);
  const long SI = (long)in_sizes[5] / D;

  char* p = (char*)d_ws;
  auto alloc = [&](size_t bytes) -> char* {
    char* r = p;
    p += (bytes + 255) & ~(size_t)255;
    return r;
  };
  unsigned short* Xbf  = (unsigned short*)alloc((size_t)T * D * 2);
  unsigned short* W13s = (unsigned short*)alloc((size_t)2 * SI * D * 2);
  unsigned short* W2s  = (unsigned short*)alloc((size_t)D * SI * 2);
  unsigned short* W13b = (unsigned short*)alloc((size_t)E * 2 * I * D * 2);
  unsigned short* W2b  = (unsigned short*)alloc((size_t)E * D * I * 2);
  unsigned short* actS = (unsigned short*)alloc((size_t)T * SI * 2);
  unsigned short* actR = (unsigned short*)alloc((size_t)T * TOPK * I * 2);
  float* yslot = (float*)alloc((size_t)T * TOPK * D * 4);
  int*   ti    = (int*)alloc((size_t)T * TOPK * 4);
  float* tw    = (float*)alloc((size_t)T * TOPK * 4);
  int* counts  = (int*)alloc(E * 4);
  int* offs    = (int*)alloc((E + 1) * 4);
  int* tlist   = (int*)alloc((size_t)T * TOPK * 4);
  int* tslot   = (int*)alloc((size_t)T * TOPK * 4);

  // 1) fp32 -> bf16 operand conversion (bandwidth pass, ~10us at 23.3 TB/s)
  const int CB = 2048;
  k_convert<<<CB, 256, 0, stream>>>(hidden, Xbf, T * D);
  k_convert<<<CB, 256, 0, stream>>>(w13s, W13s, 2 * SI * D);
  k_convert<<<CB, 256, 0, stream>>>(w2s, W2s, D * SI);
  k_convert<<<CB, 256, 0, stream>>>(w13, W13b, E * 2 * I * D);
  k_convert<<<CB, 256, 0, stream>>>(w2, W2b, E * D * I);

  // 2) routing: top-2 + deterministic per-expert compaction
  k_zero_i32<<<1, 64, 0, stream>>>(counts, (int)E);
  k_router<<<(int)((T + 255) / 256), 256, 0, stream>>>(logits, ti, tw, (int)T);
  k_count<<<(int)((T + 255) / 256), 256, 0, stream>>>(ti, counts, (int)T);
  k_prefix<<<1, 1, 0, stream>>>(counts, offs);
  k_scatter<<<(int)E, 256, 0, stream>>>(ti, offs, tlist, tslot, (int)T);

  // 3) shared expert path
  dim3 g1((unsigned)(SI / 128), (unsigned)(T / 64), 1);
  k_swiglu<<<g1, 256, 0, stream>>>(Xbf, (int)D, W13s, 0, (int)SI, actS,
                                   nullptr, nullptr, nullptr, (int)T);
  dim3 g2((unsigned)(D / 256), (unsigned)(T / 64), 1);
  k_down<<<g2, 256, 0, stream>>>(actS, (int)SI, W2s, 0, (int)D, out,
                                 nullptr, nullptr, (int)T);

  // 4) routed expert path (token-gathered; blocks past an expert's count exit)
  dim3 g3((unsigned)(I / 128), (unsigned)(T / 64), (unsigned)E);
  k_swiglu<<<g3, 256, 0, stream>>>(Xbf, (int)D, W13b, (size_t)(2 * I * D), (int)I,
                                   actR, counts, offs, tlist, (int)T);
  dim3 g4((unsigned)(D / 256), (unsigned)(T / 64), (unsigned)E);
  k_down<<<g4, 256, 0, stream>>>(actR, (int)I, W2b, (size_t)(D * I), (int)D,
                                 yslot, counts, offs, (int)T);

  // 5) weighted combine into fused output (second half of d_out)
  k_combine<<<(int)T, 256, 0, stream>>>(yslot, tslot, tw, out + (size_t)T * D, (int)D);
}
